// AttentionBlock_22428319220703
// MI455X (gfx1250) — matrix operand order repaired and checked by reading, not compiled, not run on hardware
//
#include <hip/hip_runtime.h>

// ---------------------------------------------------------------------------
// AttentionBlock for MI455X (gfx1250).
// GEMMs: v_wmma_f32_16x16x32_f16; B-operand staged in LDS via the Tensor Data
// Mover (tensor_load_to_lds + s_wait_tensorcnt), double-buffered.
// ---------------------------------------------------------------------------

typedef _Float16 h8   __attribute__((ext_vector_type(8)));   // 16B
typedef _Float16 v16h __attribute__((ext_vector_type(16)));  // 32B
typedef float    v8f  __attribute__((ext_vector_type(8)));
typedef float    f4   __attribute__((ext_vector_type(4)));

#define BB 8
#define TT 2048
#define CC 512
#define DD 512   // key/value size

// LDS staging: 64(k) x 64(n) f16 tile, row pitch padded to 72 halves (144B)
// to break bank alignment.  Two buffers for double-buffering.
#define STAGE_ROWS   64
#define STAGE_COLS   64
#define STAGE_PITCH  72                     // halves
#define STAGE_HALVES (STAGE_ROWS * STAGE_PITCH)   // 4608 halves = 9216 B

#if defined(__HIP_DEVICE_COMPILE__) && __has_builtin(__builtin_amdgcn_tensor_load_to_lds)
#define HAS_TDM 1
#else
#define HAS_TDM 0
#endif

union AFrag { v16h v; h8 h[2]; };

static __device__ inline v8f wmma_f16(v16h a, v16h b, v8f c) {
  return __builtin_amdgcn_wmma_f32_16x16x32_f16(
      /*neg_a=*/false, a, /*neg_b=*/false, b,
      /*c_mod=*/(short)0, c, /*reuse_a=*/false, /*reuse_b=*/false);
}

// A-operand fragment (16x32 MxK, f16 source, row-major, leading dim ld):
// lane L: row = m0+(L&15); halves 0..7 = K kb..kb+7, 8..15 = kb+16..kb+23,
// kb = k0 + 8*(L>>4).
static __device__ inline v16h load_a_h16(const _Float16* __restrict__ base,
                                         int ld, int m0, int k0, int lane) {
  int r  = m0 + (lane & 15);
  int kb = k0 + ((lane >> 4) << 3);
  const h8* p = (const h8*)(base + (size_t)r * ld + kb);
  AFrag f;
  f.h[0] = p[0];   // K = kb .. kb+7
  f.h[1] = p[2];   // K = kb+16 .. kb+23
  return f.v;
}

// A-operand fragment converted on the fly from fp32 source.
static __device__ inline v16h load_a_f32(const float* __restrict__ base,
                                         int ld, int m0, int k0, int lane) {
  int r  = m0 + (lane & 15);
  int kb = k0 + ((lane >> 4) << 3);
  const f4* p = (const f4*)(base + (size_t)r * ld + kb);
  f4 x0 = p[0], x1 = p[1];      // K = kb .. kb+7
  f4 y0 = p[4], y1 = p[5];      // K = kb+16 .. kb+23
  v16h o;
  o[0]=(_Float16)x0[0]; o[1]=(_Float16)x0[1]; o[2]=(_Float16)x0[2]; o[3]=(_Float16)x0[3];
  o[4]=(_Float16)x1[0]; o[5]=(_Float16)x1[1]; o[6]=(_Float16)x1[2]; o[7]=(_Float16)x1[3];
  o[8]=(_Float16)y0[0]; o[9]=(_Float16)y0[1]; o[10]=(_Float16)y0[2]; o[11]=(_Float16)y0[3];
  o[12]=(_Float16)y1[0]; o[13]=(_Float16)y1[1]; o[14]=(_Float16)y1[2]; o[15]=(_Float16)y1[3];
  return o;
}

#if HAS_TDM
typedef unsigned int u32x4 __attribute__((ext_vector_type(4)));
typedef int          i32x4 __attribute__((ext_vector_type(4)));
typedef int          i32x8 __attribute__((ext_vector_type(8)));

// Issue a TDM load of a 64x64 f16 tile (row stride ldb elements) from global
// tile-start gsrc into LDS at ldsdst, padding +16B after every 128B row.
static __device__ inline void tdm_issue(const _Float16* gsrc, int ldb,
                                        const _Float16* ldsdst) {
  unsigned long long ga = (unsigned long long)(size_t)gsrc;
  unsigned lds_off = (unsigned)(size_t)ldsdst;   // low 32 bits = LDS offset
  u32x4 g0;
  g0[0] = 1u;                                            // count=1, user mode
  g0[1] = lds_off;                                       // lds_addr
  g0[2] = (unsigned)ga;                                  // global_addr[31:0]
  g0[3] = ((unsigned)(ga >> 32) & 0x01FFFFFFu)           // global_addr[56:32]
        | 0x80000000u;                                   // type=2 ("image")
  unsigned long long st = (unsigned long long)ldb;       // dim0 stride (elems)
  const unsigned td0 = 0x7FFFFFFFu, td1 = 0x7FFFFFFFu;   // huge => no OOB clip
  i32x8 g1;
  g1[0] = (int)((1u << 16)      // data_size = 1 -> 2 bytes
              | (1u << 20)      // pad_enable
              | (4u << 22)      // pad_interval: 2^(4+3)=128B
              | (3u << 25));    // pad_amount: 4 DWORDs = 16B
  g1[1] = (int)((td0 & 0xFFFFu) << 16);                       // tensor_dim0 lo
  g1[2] = (int)(((td0 >> 16) & 0xFFFFu) | ((td1 & 0xFFFFu) << 16));
  g1[3] = (int)(((td1 >> 16) & 0xFFFFu) | ((unsigned)STAGE_COLS << 16)); // tile_dim0
  g1[4] = (int)((unsigned)STAGE_ROWS);                        // tile_dim1, tile_dim2=0
  g1[5] = (int)(st & 0xFFFFFFFFu);                            // dim0 stride lo
  g1[6] = (int)((st >> 32) & 0xFFFFu);                        // dim0 stride hi
  g1[7] = 0;
  i32x4 z4 = {0, 0, 0, 0};
  i32x8 z8 = {0, 0, 0, 0, 0, 0, 0, 0};
  // 6-arg form (this toolchain): (g0, g1, g2, g3, g4, cpol)
  __builtin_amdgcn_tensor_load_to_lds(g0, g1, z4, z4, z8, 0);
}
#endif

// One 64-deep K stage: 2 sub-steps of 32.  All four B fragments are loaded
// from LDS into registers first, then the 4 WMMAs issue back-to-back.
template <bool AF32>
static __device__ inline void compute_stage(const _Float16* __restrict__ buf,
                                            const void* __restrict__ Ap, int lda,
                                            int m0, int kbase, int lane,
                                            v8f acc[4]) {
  #pragma unroll
  for (int ks = 0; ks < 2; ++ks) {
    v16h a;
    if constexpr (AF32)
      a = load_a_f32((const float*)Ap, lda, m0, kbase + ks * 32, lane);
    else
      a = load_a_h16((const _Float16*)Ap, lda, m0, kbase + ks * 32, lane);
    const _Float16* bcol = buf + (size_t)(ks * 32 + ((lane >> 4) << 3)) * STAGE_PITCH + (lane & 15);
    AFrag bf[4];
    #pragma unroll
    for (int t = 0; t < 4; ++t) {            // column gather: K = kb..kb+7, kb+16..kb+23
      #pragma unroll
      for (int s = 0; s < 8; ++s) {
        bf[t].v[s]     = bcol[(size_t)s * STAGE_PITCH + 16 * t];
        bf[t].v[s + 8] = bcol[(size_t)(s + 16) * STAGE_PITCH + 16 * t];
      }
    }
    #pragma unroll
    for (int t = 0; t < 4; ++t)              // 4 WMMAs back-to-back
      acc[t] = wmma_f16(a, bf[t].v, acc[t]);
  }
}

// Strip GEMM: per-wave 16x64 output tile at (m0, n0); all 8 waves of the
// block share the same n-strip of B, staged in LDS (double-buffered TDM).
template <bool AF32>
static __device__ inline void gemm_strip_pipeline(
    const void* __restrict__ Ap, int lda,
    const _Float16* __restrict__ Bp, int ldb,
    int m0, int n0, int kEnd,
    _Float16* __restrict__ sm, int lane, int wave, v8f acc[4]) {
  const int S = kEnd / STAGE_ROWS;   // number of 64-deep stages
#if HAS_TDM
  if (wave == 0) tdm_issue(Bp + n0, ldb, sm);
  for (int s = 0; s < S; ++s) {
    __syncthreads();                 // everyone done reading buf[(s+1)&1]
    if (wave == 0) {
      if (s + 1 < S) {
        tdm_issue(Bp + (size_t)(s + 1) * STAGE_ROWS * ldb + n0, ldb,
                  sm + ((s + 1) & 1) * STAGE_HALVES);
        __builtin_amdgcn_s_wait_tensorcnt(1);   // stage s complete
      } else {
        __builtin_amdgcn_s_wait_tensorcnt(0);   // drain
      }
    }
    __syncthreads();                 // publish buf[s&1] to all waves
    compute_stage<AF32>(sm + (s & 1) * STAGE_HALVES, Ap, lda, m0,
                        s * STAGE_ROWS, lane, acc);
  }
#else
  // Fallback: cooperative synchronous copy into a single buffer.
  for (int s = 0; s < S; ++s) {
    __syncthreads();
    const _Float16* src = Bp + (size_t)s * STAGE_ROWS * ldb + n0;
    int tid = threadIdx.x;
    #pragma unroll
    for (int r = 0; r < 2; ++r) {
      int idx = tid + 256 * r;        // 512 h8-chunks of the 64x64 tile
      int row = idx >> 3, c8 = idx & 7;
      *(h8*)(sm + (size_t)row * STAGE_PITCH + c8 * 8) =
          *(const h8*)(src + (size_t)row * ldb + c8 * 8);
    }
    __syncthreads();
    compute_stage<AF32>(sm, Ap, lda, m0, s * STAGE_ROWS, lane, acc);
  }
#endif
}

// ---------------------------------------------------------------------------
// Kernel: transpose+convert one 512x512 fp32 weight to f16  Wt[c,n] = W[n,c]
// ---------------------------------------------------------------------------
__global__ __launch_bounds__(256) void convert_w_kernel(
    const float* __restrict__ W, _Float16* __restrict__ Wt) {
  int idx = blockIdx.x * 256 + threadIdx.x;   // 0 .. 512*512-1
  int n = idx >> 9;
  int c = idx & 511;
  Wt[(size_t)c * DD + n] = (_Float16)W[(size_t)n * CC + c];
}

// ---------------------------------------------------------------------------
// Kernel: copy x into the left half of the concat output (fp32 -> fp32)
// ---------------------------------------------------------------------------
__global__ __launch_bounds__(256) void copy_x_kernel(
    const float* __restrict__ x, float* __restrict__ out) {
  int idx4 = blockIdx.x * 256 + threadIdx.x;  // over BB*TT*CC/4 float4s
  int c4  = idx4 & 127;                       // 128 float4 per row
  int row = idx4 >> 7;
  const f4* src = (const f4*)x;
  f4* dst = (f4*)out;
  dst[(size_t)row * 256 + c4] = src[idx4];    // out row stride = 1024 floats
}

// ---------------------------------------------------------------------------
// Kernel: projection GEMM  Y[m,n] = sum_c x[m,c]*Wt[c,n] + bias[n]
//   Block: 8 waves = 128 rows x 64 cols sharing the B strip via LDS.
//   kt_layout==0: f16 row-major [BB*TT, 512]; ==1: per-batch [BB, 512, TT].
// ---------------------------------------------------------------------------
__global__ __launch_bounds__(256) void proj_kernel(
    const float* __restrict__ x, const _Float16* __restrict__ Wt,
    const float* __restrict__ bias, _Float16* __restrict__ out, int kt_layout) {
  __shared__ _Float16 smem[2 * STAGE_HALVES];
  const int lane = threadIdx.x & 31;
  const int wave = threadIdx.x >> 5;
  const int NSTRIPS = DD / 64;                   // 8
  int mb = blockIdx.x / NSTRIPS;                 // 0..127
  int ns = blockIdx.x % NSTRIPS;
  int m0 = mb * 128 + wave * 16;
  int n0 = ns * 64;

  v8f acc[4];
  #pragma unroll
  for (int t = 0; t < 4; ++t) acc[t] = (v8f){0,0,0,0,0,0,0,0};

  gemm_strip_pipeline<true>(x, CC, Wt, DD, m0, n0, CC, smem, lane, wave, acc);

  int mbase = m0 + ((lane >> 4) << 3);
  int ncol  = lane & 15;
  #pragma unroll
  for (int t = 0; t < 4; ++t) {
    int n = n0 + 16 * t + ncol;
    float bv = bias[n];
    #pragma unroll
    for (int i = 0; i < 8; ++i) {
      int m = mbase + i;
      float val = acc[t][i] + bv;
      if (kt_layout) {
        int b  = m / TT;
        int tt = m % TT;
        out[((size_t)b * DD + n) * TT + tt] = (_Float16)val;
      } else {
        out[(size_t)m * DD + n] = (_Float16)val;
      }
    }
  }
}

// ---------------------------------------------------------------------------
// Kernel: scores  S[b,q,j] = (j>q ? -inf : (q.k)/sqrt(512)), stored f16
// ---------------------------------------------------------------------------
__global__ __launch_bounds__(256) void scores_kernel(
    const _Float16* __restrict__ qh, const _Float16* __restrict__ kth,
    _Float16* __restrict__ sh) {
  __shared__ _Float16 smem[2 * STAGE_HALVES];
  const int lane = threadIdx.x & 31;
  const int wave = threadIdx.x >> 5;
  const int NSTRIPS = TT / 64;                   // 32
  const int MBLOCKS = TT / 128;                  // 16
  int b   = blockIdx.x / (NSTRIPS * MBLOCKS);
  int r   = blockIdx.x % (NSTRIPS * MBLOCKS);
  int mb  = r / NSTRIPS;
  int ns  = r % NSTRIPS;
  int m0b = mb * 128;
  int m0  = m0b + wave * 16;
  int n0  = ns * 64;

  _Float16* srow = sh + (size_t)b * TT * TT;
  int mbase = m0 + ((lane >> 4) << 3);
  int ncol  = lane & 15;

  if (n0 > m0b + 127) {                          // whole block masked
    _Float16 ninf = (_Float16)(-__builtin_inff());
    #pragma unroll
    for (int t = 0; t < 4; ++t)
      #pragma unroll
      for (int i = 0; i < 8; ++i)
        srow[(size_t)(mbase + i) * TT + (n0 + 16 * t + ncol)] = ninf;
    return;
  }

  const _Float16* qb = qh  + (size_t)b * TT * DD;
  const _Float16* kb = kth + (size_t)b * DD * TT;

  v8f acc[4];
  #pragma unroll
  for (int t = 0; t < 4; ++t) acc[t] = (v8f){0,0,0,0,0,0,0,0};

  gemm_strip_pipeline<false>(qb, DD, kb, TT, m0, n0, DD, smem, lane, wave, acc);

  const float scale = 0.04419417382415922f;      // 1/sqrt(512)
  #pragma unroll
  for (int t = 0; t < 4; ++t) {
    #pragma unroll
    for (int i = 0; i < 8; ++i) {
      int q = mbase + i;
      int j = n0 + 16 * t + ncol;
      float val = (j > q) ? -__builtin_inff() : acc[t][i] * scale;
      srow[(size_t)q * TT + j] = (_Float16)val;
    }
  }
}

// ---------------------------------------------------------------------------
// Kernel: softmax over the QUERY axis (axis=1) — one block per (b, j) column,
// in place on the f16 score buffer.
// ---------------------------------------------------------------------------
__global__ __launch_bounds__(256) void softmax_kernel(_Float16* __restrict__ sh) {
  __shared__ float red[256];
  int b = blockIdx.x / TT;
  int j = blockIdx.x % TT;
  _Float16* col = sh + (size_t)b * TT * TT + j;
  int tid = threadIdx.x;

  float v[8];
  float mx = -__builtin_inff();
  #pragma unroll
  for (int i = 0; i < 8; ++i) {
    int q = tid + 256 * i;
    v[i] = (float)col[(size_t)q * TT];
    mx = fmaxf(mx, v[i]);
  }
  red[tid] = mx; __syncthreads();
  for (int s = 128; s > 0; s >>= 1) {
    if (tid < s) red[tid] = fmaxf(red[tid], red[tid + s]);
    __syncthreads();
  }
  mx = red[0]; __syncthreads();

  float sum = 0.f;
  #pragma unroll
  for (int i = 0; i < 8; ++i) {
    v[i] = __expf(v[i] - mx);   // exp(-inf) = 0 for masked entries
    sum += v[i];
  }
  red[tid] = sum; __syncthreads();
  for (int s = 128; s > 0; s >>= 1) {
    if (tid < s) red[tid] += red[tid + s];
    __syncthreads();
  }
  float inv = 1.f / red[0];

  #pragma unroll
  for (int i = 0; i < 8; ++i)
    col[(size_t)(tid + 256 * i) * TT] = (_Float16)(v[i] * inv);
}

// ---------------------------------------------------------------------------
// Kernel: out[b,q,n] = sum_k attn[b,q,k] * v[b,k,n]; fp32 into right half of
// the concat output.  Block-level causal K bound: attn[q,k]==0 for k>q.
// ---------------------------------------------------------------------------
__global__ __launch_bounds__(256) void out_kernel(
    const _Float16* __restrict__ attn, const _Float16* __restrict__ vh,
    float* __restrict__ out) {
  __shared__ _Float16 smem[2 * STAGE_HALVES];
  const int lane = threadIdx.x & 31;
  const int wave = threadIdx.x >> 5;
  const int NSTRIPS = DD / 64;                   // 8
  const int MBLOCKS = TT / 128;                  // 16
  int b   = blockIdx.x / (NSTRIPS * MBLOCKS);
  int r   = blockIdx.x % (NSTRIPS * MBLOCKS);
  int mb  = r / NSTRIPS;
  int ns  = r % NSTRIPS;
  int m0b = mb * 128;
  int m0  = m0b + wave * 16;
  int n0  = ns * 64;

  const _Float16* ab = attn + (size_t)b * TT * TT;
  const _Float16* vb = vh   + (size_t)b * TT * DD;

  v8f acc[4];
  #pragma unroll
  for (int t = 0; t < 4; ++t) acc[t] = (v8f){0,0,0,0,0,0,0,0};

  int kEnd = m0b + 128;                          // causal block bound
  gemm_strip_pipeline<false>(ab, TT, vb, DD, m0, n0, kEnd, smem, lane, wave, acc);

  int mbase = m0 + ((lane >> 4) << 3);
  int ncol  = lane & 15;
  #pragma unroll
  for (int t = 0; t < 4; ++t) {
    int n = n0 + 16 * t + ncol;
    #pragma unroll
    for (int i = 0; i < 8; ++i) {
      int q = mbase + i;
      out[((size_t)b * TT + q) * 1024 + 512 + n] = acc[t][i];
    }
  }
}

// ---------------------------------------------------------------------------
// Host launcher
// ---------------------------------------------------------------------------
extern "C" void kernel_launch(void* const* d_in, const int* in_sizes, int n_in,
                              void* d_out, int out_size, void* d_ws, size_t ws_size,
                              hipStream_t stream) {
  const float* x  = (const float*)d_in[0];
  const float* Wq = (const float*)d_in[1];
  const float* bq = (const float*)d_in[2];
  const float* Wk = (const float*)d_in[3];
  const float* bk = (const float*)d_in[4];
  const float* Wv = (const float*)d_in[5];
  const float* bv = (const float*)d_in[6];
  float* out = (float*)d_out;

  // Workspace layout (bytes)
  char* ws = (char*)d_ws;
  const size_t W_BYTES   = (size_t)CC * DD * sizeof(_Float16);       // 512 KB
  const size_t QKV_BYTES = (size_t)BB * TT * DD * sizeof(_Float16);  // 16 MB
  _Float16* wtq = (_Float16*)(ws);
  _Float16* wtk = (_Float16*)(ws + W_BYTES);
  _Float16* wtv = (_Float16*)(ws + 2 * W_BYTES);
  _Float16* qh  = (_Float16*)(ws + 3 * W_BYTES);
  _Float16* kth = (_Float16*)(ws + 3 * W_BYTES + QKV_BYTES);
  _Float16* vh  = (_Float16*)(ws + 3 * W_BYTES + 2 * QKV_BYTES);
  _Float16* sh  = (_Float16*)(ws + 3 * W_BYTES + 3 * QKV_BYTES);     // 64 MB

  // 1) weight transpose+convert (512x512 each)
  convert_w_kernel<<<1024, 256, 0, stream>>>(Wq, wtq);
  convert_w_kernel<<<1024, 256, 0, stream>>>(Wk, wtk);
  convert_w_kernel<<<1024, 256, 0, stream>>>(Wv, wtv);

  // 2) x passthrough into left half of output
  copy_x_kernel<<<(BB * TT * CC / 4) / 256, 256, 0, stream>>>(x, out);

  // 3) projections: q row-major, k transposed [B,512,T], v row-major
  proj_kernel<<<1024, 256, 0, stream>>>(x, wtq, bq, qh, 0);
  proj_kernel<<<1024, 256, 0, stream>>>(x, wtk, bk, kth, 1);
  proj_kernel<<<1024, 256, 0, stream>>>(x, wtv, bv, vh, 0);

  // 4) masked, scaled scores (f16 in ws)
  scores_kernel<<<4096, 256, 0, stream>>>(qh, kth, sh);

  // 5) softmax over query axis, in place
  softmax_kernel<<<BB * TT, 256, 0, stream>>>(sh);

  // 6) attn @ V into right half of output
  out_kernel<<<1024, 256, 0, stream>>>(sh, vh, out);
}
